// H_ACS_Encoder_86732569575515
// MI455X (gfx1250) — matrix operand order repaired and checked by reading, not compiled, not run on hardware
//
#include <hip/hip_runtime.h>

// ---------------------------------------------------------------------------
// Types for WMMA fragments (gfx1250, wave32)
// ---------------------------------------------------------------------------
typedef __attribute__((ext_vector_type(16))) __bf16     v16bf;
typedef __attribute__((ext_vector_type(8)))  float      v8f;
typedef __attribute__((ext_vector_type(4)))  unsigned   u32x4;
typedef __attribute__((ext_vector_type(4)))  int        i32x4;

union Frag {
  v16bf v;
  u32x4 q[2];
};

#define AS1G __attribute__((address_space(1)))
#define AS3L __attribute__((address_space(3)))

#if defined(__has_builtin)
#  if __has_builtin(__builtin_amdgcn_global_load_async_to_lds_b128)
#    define USE_ASYNC_LDS 1
#  else
#    define USE_ASYNC_LDS 0
#  endif
#else
#  define USE_ASYNC_LDS 0
#endif

// 16-byte global -> LDS copy; async (ASYNCcnt-tracked) when available.
__device__ __forceinline__ void copy16_g2l(const u32x4* g, u32x4* l) {
#if USE_ASYNC_LDS
  __builtin_amdgcn_global_load_async_to_lds_b128((AS1G i32x4*)g,
                                                 (AS3L i32x4*)l, 0, 0);
#else
  *l = *g;
#endif
}

__device__ __forceinline__ void wait_async_copies() {
#if USE_ASYNC_LDS
#  if __has_builtin(__builtin_amdgcn_s_wait_asynccnt)
  __builtin_amdgcn_s_wait_asynccnt(0);
#  else
  asm volatile("s_wait_asynccnt 0x0" ::: "memory");
#  endif
#endif
}

__device__ __forceinline__ unsigned short f32_to_bf16(float f) {
  unsigned u = __float_as_uint(f);
  unsigned r = 0x7FFFu + ((u >> 16) & 1u);   // round-to-nearest-even
  return (unsigned short)((u + r) >> 16);
}

// monotonic float -> uint key (for exact top-k threshold search)
__device__ __forceinline__ unsigned order_key(float f) {
  unsigned u = __float_as_uint(f);
  unsigned mask = (unsigned)(((int)u) >> 31) | 0x80000000u;
  return u ^ mask;
}

// ---------------------------------------------------------------------------
// Elementwise f32 -> bf16 convert (for W)
// ---------------------------------------------------------------------------
__global__ void cvt_bf16_kernel(const float* __restrict__ in,
                                unsigned short* __restrict__ out, int n) {
  int i = blockIdx.x * blockDim.x + threadIdx.x;
  if (i < n) out[i] = f32_to_bf16(in[i]);
}

// ---------------------------------------------------------------------------
// Row l2-normalize -> bf16 (for atom bank, and for P -> q)
// ---------------------------------------------------------------------------
__global__ __launch_bounds__(256)
void rownorm_bf16_kernel(const float* __restrict__ in,
                         unsigned short* __restrict__ out, int cols) {
  const int tid = threadIdx.x;
  const size_t row = blockIdx.x;
  const float* r = in + row * (size_t)cols;
  __shared__ float red[256];
  float s = 0.f;
  for (int c = tid; c < cols; c += 256) { float v = r[c]; s += v * v; }
  red[tid] = s; __syncthreads();
  for (int off = 128; off > 0; off >>= 1) {
    if (tid < off) red[tid] += red[tid + off];
    __syncthreads();
  }
  float inv = 1.0f / fmaxf(sqrtf(red[0]), 1e-12f);
  for (int c = tid; c < cols; c += 256)
    out[row * (size_t)cols + c] = f32_to_bf16(r[c] * inv);
}

// ---------------------------------------------------------------------------
// base_emb = l2norm(base_raw) (f32 out)  +  bf16 copy of raw base for GEMM1
// ---------------------------------------------------------------------------
__global__ __launch_bounds__(256)
void base_prep_kernel(const float* __restrict__ in,
                      float* __restrict__ out_norm,
                      unsigned short* __restrict__ out_bf_raw, int cols) {
  const int tid = threadIdx.x;
  const size_t row = blockIdx.x;
  const float* r = in + row * (size_t)cols;
  __shared__ float red[256];
  float s = 0.f;
  for (int c = tid; c < cols; c += 256) { float v = r[c]; s += v * v; }
  red[tid] = s; __syncthreads();
  for (int off = 128; off > 0; off >>= 1) {
    if (tid < off) red[tid] += red[tid + off];
    __syncthreads();
  }
  float inv = 1.0f / fmaxf(sqrtf(red[0]), 1e-12f);
  for (int c = tid; c < cols; c += 256) {
    float v = r[c];
    __builtin_nontemporal_store(v * inv, &out_norm[row * (size_t)cols + c]);
    out_bf_raw[row * (size_t)cols + c] = f32_to_bf16(v);
  }
}

// ---------------------------------------------------------------------------
// C[M,N] = A[M,K] * B[N,K]^T (+bias), bf16 in, f32 acc, WMMA 16x16x32.
// Block = 256 thr = 8 wave32; tile 64(M) x 128(N); K-step 64.
// LDS double buffer: A 64x64 bf16 + B 128x64 bf16, rows padded to 144B
// (stride 9 u32x4) => 16-lane ds_load_b128 fragment reads are
// bank-conflict-free. Tile kt+1 streams via GLOBAL_LOAD_ASYNC_TO_LDS_B128
// (ASYNCcnt) while WMMAs consume tile kt. All 20 fragment ds_loads for a
// K-step are issued before the 8 back-to-back WMMAs so the ds latency is
// covered by one batched wait instead of per-WMMA s_wait_dscnt 0.
// Requires M%64==0, N%128==0, K%64==0 (holds for all uses here).
// ---------------------------------------------------------------------------
#define A_UNITS 576            // 64 rows * 9 u32x4
#define B_UNITS 1152           // 128 rows * 9 u32x4
#define BUF_UNITS (A_UNITS + B_UNITS)

__global__ __launch_bounds__(256)
void gemm_nt_bf16_kernel(const unsigned short* __restrict__ Am,
                         const unsigned short* __restrict__ Bm,
                         float* __restrict__ C,
                         const float* __restrict__ bias,
                         int M, int N, int K) {
  __shared__ u32x4 smem[2 * BUF_UNITS];   // 55296 B

  const int tid  = threadIdx.x;
  const int lane = tid & 31;
  const int wave = tid >> 5;
  const int mt = wave & 3;                // wave's 16-row strip in the 64-row tile
  const int nh = wave >> 2;               // wave's 64-col half of the 128-col tile
  const int lm = lane & 15;
  const int kh = lane >> 4;

  const int m0b = blockIdx.y * 64;
  const int n0b = blockIdx.x * 128;

  // ---- tile copy: 6 x 16B per thread (2 for A, 4 for B) ----
  auto issueTile = [&](int k0, int buf) {
    u32x4* bA = smem + buf * BUF_UNITS;
    u32x4* bB = bA + A_UNITS;
#pragma unroll
    for (int i = 0; i < 2; ++i) {
      int c = tid + 256 * i;              // 512 chunks for A (64 rows x 8)
      int row = c >> 3, kc = c & 7;       // kc: 16B chunk within row
      copy16_g2l((const u32x4*)(Am + (size_t)(m0b + row) * K + k0 + kc * 8),
                 bA + row * 9 + kc);
    }
#pragma unroll
    for (int i = 0; i < 4; ++i) {
      int c = tid + 256 * i;              // 1024 chunks for B (128 rows x 8)
      int row = c >> 3, kc = c & 7;
      copy16_g2l((const u32x4*)(Bm + (size_t)(n0b + row) * K + k0 + kc * 8),
                 bB + row * 9 + kc);
    }
  };

  v8f acc[4] = {v8f{}, v8f{}, v8f{}, v8f{}};

  const int arow9 = (mt * 16 + lm) * 9;   // A fragment row base (u32x4 units)
  int brow9[4];
#pragma unroll
  for (int j = 0; j < 4; ++j) brow9[j] = (nh * 64 + j * 16 + lm) * 9;

  const int nk = K >> 6;                  // K/64 tiles
  issueTile(0, 0);
  wait_async_copies();
  __syncthreads();

  for (int kt = 0; kt < nk; ++kt) {
    const int buf = kt & 1;
    if (kt + 1 < nk) issueTile((kt + 1) << 6, buf ^ 1);

    const u32x4* bA = smem + buf * BUF_UNITS;
    const u32x4* bB = bA + A_UNITS;

    // Load ALL fragments for this 64-deep K-step first (20 x ds_load_b128),
    // then run the 8 WMMAs back-to-back.
    Frag afr[2];
    Frag bfr[2][4];
#pragma unroll
    for (int s = 0; s < 2; ++s) {
      const int ks8 = s * 4;               // ks/8 (ks = 0, 32)
      const int ai = arow9 + ks8 + kh;
      afr[s].q[0] = bA[ai];                // K = ks + kh*8 + [0..7]
      afr[s].q[1] = bA[ai + 2];            // K = ks + 16 + kh*8 + [0..7]
#pragma unroll
      for (int j = 0; j < 4; ++j) {
        const int bi = brow9[j] + ks8 + kh;
        bfr[s][j].q[0] = bB[bi];           // K = ks + kh*8 + [0..7]
        bfr[s][j].q[1] = bB[bi + 2];       // K = ks + 16 + kh*8 + [0..7]
      }
    }
#pragma unroll
    for (int s = 0; s < 2; ++s) {
#pragma unroll
      for (int j = 0; j < 4; ++j) {
        acc[j] = __builtin_amdgcn_wmma_f32_16x16x32_bf16(
            false, afr[s].v, false, bfr[s][j].v, (short)0, acc[j], false, false);
      }
    }

    __syncthreads();                       // done reading buf before it is refilled
    if (kt + 1 < nk) {
      wait_async_copies();                 // tile kt+1 landed
      __syncthreads();
    }
  }

  // C/D layout: VGPR r -> row m0 + kh*8 + r, col = n-tile + (lane&15)
  const int ms = m0b + mt * 16 + kh * 8;
#pragma unroll
  for (int j = 0; j < 4; ++j) {
    const int col = n0b + nh * 64 + j * 16 + lm;
    const float bv = bias ? bias[col] : 0.0f;
#pragma unroll
    for (int r = 0; r < 8; ++r)
      C[(size_t)(ms + r) * N + col] = acc[j][r] + bv;
  }
}

// ---------------------------------------------------------------------------
// Per-row: exact top-k (radix bisection on order keys, tie cap via
// ds_add_rtn), softmax(T=0.7), dense scatter, sparse reconstruction
// euc_raw = sum w_j * atom[j] (atom_bank is L2-resident), euc_vec = l2norm.
// ---------------------------------------------------------------------------
__global__ __launch_bounds__(256)
void topk_kernel(const float* __restrict__ scores,
                 const float* __restrict__ atoms,
                 const int* __restrict__ topk_p,
                 float* __restrict__ weights,
                 float* __restrict__ euc_raw,
                 float* __restrict__ euc_vec,
                 int A, int D) {
  __shared__ float    s_sc[4096];
  __shared__ float    s_redf[256];
  __shared__ unsigned s_redu[256];
  __shared__ int      s_selidx[128];
  __shared__ float    s_selw[128];
  __shared__ int      s_cnt[2];

  const int tid = threadIdx.x;
  const size_t row = blockIdx.x;
  int k = topk_p ? topk_p[0] : 128;
  k = k < 1 ? 1 : (k > 128 ? 128 : k);

  float mymax = -3.4e38f;
  for (int i = tid; i < A; i += 256) {
    float v = scores[row * (size_t)A + i];
    s_sc[i] = v;
    mymax = fmaxf(mymax, v);
  }
  s_redf[tid] = mymax; __syncthreads();
  for (int off = 128; off > 0; off >>= 1) {
    if (tid < off) s_redf[tid] = fmaxf(s_redf[tid], s_redf[tid + off]);
    __syncthreads();
  }
  const float smax = s_redf[0];
  __syncthreads();

  // exact k-th largest key: max thr with count(key >= thr) >= k
  unsigned thr = 0u;
  for (int bit = 31; bit >= 0; --bit) {
    unsigned t = thr | (1u << bit);
    unsigned c = 0;
    for (int i = tid; i < A; i += 256) c += (order_key(s_sc[i]) >= t);
    s_redu[tid] = c; __syncthreads();
    for (int off = 128; off > 0; off >>= 1) {
      if (tid < off) s_redu[tid] += s_redu[tid + off];
      __syncthreads();
    }
    if (s_redu[0] >= (unsigned)k) thr = t;
    __syncthreads();
  }

  unsigned cg = 0;
  for (int i = tid; i < A; i += 256) cg += (order_key(s_sc[i]) > thr);
  s_redu[tid] = cg; __syncthreads();
  for (int off = 128; off > 0; off >>= 1) {
    if (tid < off) s_redu[tid] += s_redu[tid + off];
    __syncthreads();
  }
  const int nEqual = k - (int)s_redu[0];
  if (tid == 0) { s_cnt[0] = 0; s_cnt[1] = 0; }
  __syncthreads();

  const float invT = 1.0f / 0.7f;
  float mysum = 0.f;
  for (int i = tid; i < A; i += 256) {
    const unsigned key = order_key(s_sc[i]);
    bool sel = key > thr;
    if (!sel && key == thr) {
      int old = atomicAdd(&s_cnt[0], 1);           // ds_add_rtn_u32 tie cap
      sel = (old < nEqual);
    }
    float w = 0.f;
    if (sel) {
      w = __expf((s_sc[i] - smax) * invT);
      int pos = atomicAdd(&s_cnt[1], 1);
      if (pos < 128) { s_selidx[pos] = i; s_selw[pos] = w; }
      mysum += w;
    }
    s_sc[i] = w;
  }
  s_redf[tid] = mysum; __syncthreads();
  for (int off = 128; off > 0; off >>= 1) {
    if (tid < off) s_redf[tid] += s_redf[tid + off];
    __syncthreads();
  }
  const float invw = 1.0f / s_redf[0];

  for (int i = tid; i < A; i += 256)
    __builtin_nontemporal_store(s_sc[i] * invw, &weights[row * (size_t)A + i]);

  const int ksel = s_cnt[1] < 128 ? s_cnt[1] : 128;
  if (tid < ksel) s_selw[tid] *= invw;
  __syncthreads();

  float acc[4] = {0.f, 0.f, 0.f, 0.f};
  for (int j = 0; j < ksel; ++j) {
    const float* ar = atoms + (size_t)s_selidx[j] * D;
    const float w = s_selw[j];
#pragma unroll
    for (int c = 0; c < 4; ++c) {
      int col = tid + c * 256;
      if (col < D) acc[c] = fmaf(w, ar[col], acc[c]);
    }
  }
  float ss = 0.f;
#pragma unroll
  for (int c = 0; c < 4; ++c) ss += acc[c] * acc[c];
  __syncthreads();
  s_redf[tid] = ss; __syncthreads();
  for (int off = 128; off > 0; off >>= 1) {
    if (tid < off) s_redf[tid] += s_redf[tid + off];
    __syncthreads();
  }
  const float invn = 1.0f / fmaxf(sqrtf(s_redf[0]), 1e-12f);
#pragma unroll
  for (int c = 0; c < 4; ++c) {
    int col = tid + c * 256;
    if (col < D) {
      __builtin_nontemporal_store(acc[c],        &euc_raw[row * (size_t)D + col]);
      __builtin_nontemporal_store(acc[c] * invn, &euc_vec[row * (size_t)D + col]);
    }
  }
}

// ---------------------------------------------------------------------------
// Host launcher
// ---------------------------------------------------------------------------
extern "C" void kernel_launch(void* const* d_in, const int* in_sizes, int n_in,
                              void* d_out, int out_size, void* d_ws, size_t ws_size,
                              hipStream_t stream) {
  const int D = in_sizes[2];              // 1024 (bias length)
  const int B = in_sizes[0] / D;          // 16384
  const int A = in_sizes[3] / D;          // 4096

  const float* base_raw = (const float*)d_in[0];
  const float* Wm       = (const float*)d_in[1];
  const float* bias     = (const float*)d_in[2];
  const float* atoms    = (const float*)d_in[3];
  const int*   topk     = (const int*)d_in[4];

  float* out_base = (float*)d_out;                       // [B,D] base_emb
  float* out_w    = out_base + (size_t)B * D;            // [B,A] weights
  float* out_sc   = out_w    + (size_t)B * A;            // [B,A] scores
  float* out_er   = out_sc   + (size_t)B * A;            // [B,D] euc_raw
  float* out_ev   = out_er   + (size_t)B * D;            // [B,D] euc_vec

  char* w = (char*)d_ws;
  float* P                 = (float*)w;          w += (size_t)B * D * 4; // GEMM1 out
  unsigned short* base_bf  = (unsigned short*)w; w += (size_t)B * D * 2;
  unsigned short* q_bf     = (unsigned short*)w; w += (size_t)B * D * 2;
  unsigned short* a_bf     = (unsigned short*)w; w += (size_t)A * D * 2;
  unsigned short* W_bf     = (unsigned short*)w;

  // 1) W -> bf16
  cvt_bf16_kernel<<<(D * D + 255) / 256, 256, 0, stream>>>(Wm, W_bf, D * D);
  // 2) a = l2norm(atom_bank) -> bf16
  rownorm_bf16_kernel<<<A, 256, 0, stream>>>(atoms, a_bf, D);
  // 3) base_emb (f32) + raw base -> bf16
  base_prep_kernel<<<B, 256, 0, stream>>>(base_raw, out_base, base_bf, D);
  // 4) P = base @ W^T + b   (WMMA bf16, f32 acc, async-LDS pipelined)
  dim3 g1(D / 128, B / 64);
  gemm_nt_bf16_kernel<<<g1, 256, 0, stream>>>(base_bf, W_bf, P, bias, B, D, D);
  // 5) q = l2norm(P) -> bf16
  rownorm_bf16_kernel<<<B, 256, 0, stream>>>(P, q_bf, D);
  // 6) scores = q @ a^T   (WMMA bf16, f32 acc, async-LDS pipelined)
  dim3 g2(A / 128, B / 64);
  gemm_nt_bf16_kernel<<<g2, 256, 0, stream>>>(q_bf, a_bf, out_sc, nullptr, B, A, D);
  // 7) top-k softmax + dense weights + sparse reconstruction + l2norm
  topk_kernel<<<B, 256, 0, stream>>>(out_sc, atoms, topk, out_w, out_er, out_ev, A, D);
}
